// involution_5669356834469
// MI455X (gfx1250) — compile-verified, matrix operand order checked by reading, not run on hardware
//
#include <hip/hip_runtime.h>

// ---------------------------------------------------------------------------
// Involution for MI455X (gfx1250, wave32, WMMA f32_16x16x32_f16)
//   x:  (4,256,56,56) f32     w1: (64,256)    bn*: (64)
//   w2: (784,64)              b2: (784)       out: (4,256,56,56) f32
//
// All WMMA operands stored K-contiguous per row; fragments are two 16B loads
// with the interleaved K pattern from the 16-bit A-matrix table (CDNA5 ISA):
//   lane<16: K = {k0..k0+7, k0+16..k0+23} ; lane>=16: K = {k0+8..15, k0+24..31}
// ---------------------------------------------------------------------------

typedef _Float16 h8   __attribute__((ext_vector_type(8)));
typedef _Float16 v16h __attribute__((ext_vector_type(16)));
typedef float    v8f  __attribute__((ext_vector_type(8)));
typedef float    f4   __attribute__((ext_vector_type(4)));

union V16U { v16h v; h8 h[2]; };

#define HWPIX 3136     // 56*56
#define CIN   256
#define CR    64
#define O2    784      // 49 * 16

// ws layout (bytes)
#define W1H_OFF  0            //  64*256*2  =   32,768
#define W2H_OFF  65536        // 784*64*2   =  100,352
#define XT_OFF   262144       // 4*3136*256*2 = 6,422,528  (x transposed [b][hw][c] f16)
#define T_OFF    6684672      // 4*3136*64*2  = 1,605,632  (t [b][hw][cr] f16)

// ---------------------------------------------------------------------------
// Kernel 0a: convert w1 and w2 to f16 (layouts already K-contiguous)
// ---------------------------------------------------------------------------
__global__ void conv_w_kernel(const float* __restrict__ w1, const float* __restrict__ w2,
                              _Float16* __restrict__ w1h, _Float16* __restrict__ w2h) {
    int i = blockIdx.x * blockDim.x + threadIdx.x;
    if (i < CR * CIN)            w1h[i] = (_Float16)w1[i];
    else if (i < CR*CIN + O2*CR) w2h[i - CR*CIN] = (_Float16)w2[i - CR*CIN];
}

// ---------------------------------------------------------------------------
// Kernel 0b: transpose+convert x -> xh_t[b][hw][c] f16, tiled through LDS
//   grid = B*49 (64-pixel tiles), 256 threads
// ---------------------------------------------------------------------------
__launch_bounds__(256)
__global__ void transpose_x_kernel(const float* __restrict__ x, _Float16* __restrict__ xh_t) {
    __shared__ _Float16 tile[64][264];   // [pixel][channel], padded pitch

    const int tid = threadIdx.x;
    const int b   = blockIdx.x / 49;
    const int hw0 = (blockIdx.x % 49) * 64;
    const float* xb = x + (size_t)b * CIN * HWPIX + hw0;

    // load: coalesced float4 per thread, scatter-transpose into LDS
    #pragma unroll
    for (int it = 0; it < 16; ++it) {
        int idx = tid + 256 * it;        // 4096 float4 chunks = 256 rows x 16
        int c   = idx >> 4;
        int col = (idx & 15) * 4;
        f4 v = *(const f4*)(xb + (size_t)c * HWPIX + col);
        tile[col + 0][c] = (_Float16)v.x;
        tile[col + 1][c] = (_Float16)v.y;
        tile[col + 2][c] = (_Float16)v.z;
        tile[col + 3][c] = (_Float16)v.w;
    }
    __syncthreads();

    // store: coalesced 16B rows of xh_t[hw][c]
    _Float16* ob = xh_t + ((size_t)b * HWPIX + hw0) * CIN;
    #pragma unroll
    for (int it = 0; it < 8; ++it) {
        int idx = tid + 256 * it;        // 2048 h8 chunks = 64 rows x 32
        int r   = idx >> 5;
        int seg = (idx & 31) * 8;
        *(h8*)(ob + (size_t)r * CIN + seg) = *(const h8*)&tile[r][seg];
    }
}

// ---------------------------------------------------------------------------
// Kernel 1: t[b][hw][cr] = f16( relu( BN( w1 @ x ) ) )  -- LDS-free WMMA
//   grid = B*49 (64-pixel tiles), 256 threads = 8 waves
//   wave -> (mt = wv>>1 of 4 M-tiles, two N-subtiles)
// ---------------------------------------------------------------------------
__launch_bounds__(256)
__global__ void gen_t_kernel(const _Float16* __restrict__ w1h,
                             const float* __restrict__ gamma, const float* __restrict__ beta,
                             const float* __restrict__ mean,  const float* __restrict__ var,
                             const _Float16* __restrict__ xh_t,
                             _Float16* __restrict__ t_ws)
{
    __shared__ float sc[64], sh[64];

    const int tid = threadIdx.x;
    const int b   = blockIdx.x / 49;
    const int hw0 = (blockIdx.x % 49) * 64;

    if (tid < 64) {
        float s = gamma[tid] * rsqrtf(var[tid] + 1e-5f);
        sc[tid] = s;
        sh[tid] = beta[tid] - mean[tid] * s;
    }
    __syncthreads();

    const int lane = tid & 31;
    const int wv   = tid >> 5;
    const int mt   = wv >> 1;
    const int nt0  = (wv & 1) * 2;
    const int l16  = lane & 15;
    const int khi  = lane >> 4;

    const _Float16* arow  = w1h + (size_t)(mt * 16 + l16) * CIN + khi * 8;
    const _Float16* brow0 = xh_t + ((size_t)b * HWPIX + hw0 + nt0 * 16 + l16) * CIN + khi * 8;
    const _Float16* brow1 = brow0 + (size_t)16 * CIN;

    v8f acc0 = {}, acc1 = {};
    #pragma unroll
    for (int k0 = 0; k0 < 256; k0 += 32) {
        V16U a, b0, b1;
        a.h[0]  = *(const h8*)(arow  + k0);
        a.h[1]  = *(const h8*)(arow  + k0 + 16);
        b0.h[0] = *(const h8*)(brow0 + k0);
        b0.h[1] = *(const h8*)(brow0 + k0 + 16);
        b1.h[0] = *(const h8*)(brow1 + k0);
        b1.h[1] = *(const h8*)(brow1 + k0 + 16);
        acc0 = __builtin_amdgcn_wmma_f32_16x16x32_f16(false, a.v, false, b0.v, (short)0, acc0, false, false);
        acc1 = __builtin_amdgcn_wmma_f32_16x16x32_f16(false, a.v, false, b1.v, (short)0, acc1, false, false);
    }

    // BN + ReLU; D rows per lane are 8 contiguous cr -> one packed 16B store each
    const int basem = khi * 8;
    #pragma unroll
    for (int half = 0; half < 2; ++half) {
        v8f acc = half ? acc1 : acc0;
        int hw = hw0 + (nt0 + half) * 16 + l16;
        h8 outv;
        #pragma unroll
        for (int v = 0; v < 8; ++v) {
            int cr = mt * 16 + basem + v;
            float f = acc[v] * sc[cr] + sh[cr];
            outv[v] = (_Float16)fmaxf(f, 0.0f);
        }
        *(h8*)(t_ws + ((size_t)b * HWPIX + hw) * CR + mt * 16 + basem) = outv;
    }
}

// ---------------------------------------------------------------------------
// Kernel 2: fused  wgt = w2 @ t + b2  (WMMA -> LDS tile)  ->  involution
//   grid = B*98 (32-pixel tiles), 256 threads, dynamic LDS = 784*33*4 B
// ---------------------------------------------------------------------------
#define WGPITCH 33

__launch_bounds__(256)
__global__ void invol_kernel(const float* __restrict__ x, const _Float16* __restrict__ w2h,
                             const float* __restrict__ b2, const _Float16* __restrict__ t_ws,
                             float* __restrict__ out)
{
    extern __shared__ float wg[];        // [784][33]

    const int tid  = threadIdx.x;
    const int b    = blockIdx.x / 98;
    const int hw0  = (blockIdx.x % 98) * 32;
    const int lane = tid & 31;
    const int wv   = tid >> 5;
    const int l16  = lane & 15;
    const int khi  = lane >> 4;

    const float* xb = x + (size_t)b * CIN * HWPIX;

    // warm the involution window while the GEMM runs (global_prefetch_b8)
    {
        int pf0 = hw0 >= 168 ? hw0 - 168 : 0;
        const float* pc = xb + (size_t)tid * HWPIX;
        __builtin_prefetch(pc + pf0, 0, 0);
        __builtin_prefetch(pc + hw0, 0, 0);
        __builtin_prefetch(pc + hw0 + 168 + 31, 0, 0);
    }

    // GEMM2: wg[o][p] = w2[o][:] . t[:,p] + b2[o],  M=784 (49 tiles) K=64 N=32
    const _Float16* tb  = t_ws + ((size_t)b * HWPIX + hw0) * CR + khi * 8;
    const _Float16* b0p = tb + (size_t)l16 * CR;
    const _Float16* b1p = tb + (size_t)(16 + l16) * CR;

    for (int mtile = wv; mtile < 49; mtile += 8) {
        v8f acc0 = {}, acc1 = {};
        const _Float16* ap = w2h + (size_t)(mtile * 16 + l16) * CR + khi * 8;
        #pragma unroll
        for (int ck = 0; ck < 2; ++ck) {
            int k0 = ck * 32;
            V16U a, bb0, bb1;
            a.h[0]   = *(const h8*)(ap  + k0);
            a.h[1]   = *(const h8*)(ap  + k0 + 16);
            bb0.h[0] = *(const h8*)(b0p + k0);
            bb0.h[1] = *(const h8*)(b0p + k0 + 16);
            bb1.h[0] = *(const h8*)(b1p + k0);
            bb1.h[1] = *(const h8*)(b1p + k0 + 16);
            acc0 = __builtin_amdgcn_wmma_f32_16x16x32_f16(false, a.v, false, bb0.v, (short)0, acc0, false, false);
            acc1 = __builtin_amdgcn_wmma_f32_16x16x32_f16(false, a.v, false, bb1.v, (short)0, acc1, false, false);
        }
        int basem = khi * 8;
        #pragma unroll
        for (int v = 0; v < 8; ++v) {
            int o = mtile * 16 + basem + v;
            float bias = b2[o];
            wg[o * WGPITCH + l16]      = acc0[v] + bias;
            wg[o * WGPITCH + 16 + l16] = acc1[v] + bias;
        }
    }
    __syncthreads();

    // Involution: lane = pixel, wave handles 2 groups (32 channels).
    // Per-pixel 49 weights shared by the 16 channels of a group -> registers.
    const int p   = lane;
    const int hw  = hw0 + p;
    const int hh0 = hw / 56;
    const int ww0 = hw % 56;
    float* ob = out + (size_t)b * CIN * HWPIX;

    #pragma unroll
    for (int gg = 0; gg < 2; ++gg) {
        int g = wv * 2 + gg;
        float wk[49];
        #pragma unroll
        for (int kk = 0; kk < 49; ++kk)
            wk[kk] = wg[(g * 49 + kk) * WGPITCH + p];

        for (int jc = 0; jc < 16; ++jc) {
            int c = g * 16 + jc;
            const float* xc = xb + (size_t)c * HWPIX;
            float acc = 0.0f;
            #pragma unroll
            for (int di = 0; di < 7; ++di) {
                int hh = hh0 + di - 3;
                bool vrow = (hh >= 0) & (hh < 56);
                const float* xr = xc + hh * 56;
                #pragma unroll
                for (int dj = 0; dj < 7; ++dj) {
                    int ww = ww0 + dj - 3;
                    float xv = (vrow && ww >= 0 && ww < 56) ? xr[ww] : 0.0f;
                    acc = fmaf(xv, wk[di * 7 + dj], acc);
                }
            }
            ob[(size_t)c * HWPIX + hw] = acc;
        }
    }
}

// ---------------------------------------------------------------------------
extern "C" void kernel_launch(void* const* d_in, const int* in_sizes, int n_in,
                              void* d_out, int out_size, void* d_ws, size_t ws_size,
                              hipStream_t stream) {
    const float* x     = (const float*)d_in[0];
    const float* w1    = (const float*)d_in[1];
    const float* gamma = (const float*)d_in[2];
    const float* beta  = (const float*)d_in[3];
    const float* mean  = (const float*)d_in[4];
    const float* var   = (const float*)d_in[5];
    const float* w2    = (const float*)d_in[6];
    const float* b2    = (const float*)d_in[7];
    float* out = (float*)d_out;

    _Float16* w1h  = (_Float16*)((char*)d_ws + W1H_OFF);
    _Float16* w2h  = (_Float16*)((char*)d_ws + W2H_OFF);
    _Float16* xh_t = (_Float16*)((char*)d_ws + XT_OFF);
    _Float16* t_ws = (_Float16*)((char*)d_ws + T_OFF);

    conv_w_kernel<<<(CR*CIN + O2*CR + 255) / 256, 256, 0, stream>>>(w1, w2, w1h, w2h);
    transpose_x_kernel<<<4 * 49, 256, 0, stream>>>(x, xh_t);
    gen_t_kernel<<<4 * 49, 256, 0, stream>>>(w1h, gamma, beta, mean, var, xh_t, t_ws);

    size_t smem2 = (size_t)O2 * WGPITCH * 4;   // ~103.5 KB < 320 KB/WGP
    invol_kernel<<<4 * 98, 256, smem2, stream>>>(x, w2h, b2, t_ws, out);
}
